// LocalAttention_62955630625180
// MI455X (gfx1250) — compile-verified, matrix operand order checked by reading, not run on hardware
//
#include <hip/hip_runtime.h>
#include <stdint.h>

typedef __attribute__((ext_vector_type(16))) _Float16 v16h;
typedef __attribute__((ext_vector_type(8)))  float    v8f;

#define S_LEN 4096
#define T_LEN 2048
#define B_SZ  4
#define DQ    1024
#define DC    1024
#define NROWS (B_SZ * T_LEN)   /* 8192 rows of (b,t) */
#define PPAD  112
#define PSIZE 100
#define LDSTRIDE 40            /* halves per row in LDS tiles: 32 data + 8 pad (80B) */

union Frag { uint4 u[2]; v16h h; };

static __device__ __forceinline__ v8f wmma_f16(v16h a, v16h b, v8f c) {
  // D = A(16x32 f16) * B(32x16 f16) + C(16x16 f32)
  return __builtin_amdgcn_wmma_f32_16x16x32_f16(
      /*neg_a=*/false, a, /*neg_b=*/false, b,
      /*c_mod=*/(short)0, c, /*reuse_a=*/false, /*reuse_b=*/false);
}

// Async global->LDS copy of 32 contiguous bytes (2 x b128). INST_OFFSET is
// added to BOTH the LDS and the global address (ISA 15.18 async pseudocode),
// so one address pair covers both chunks. Tracked by ASYNCcnt.
static __device__ __forceinline__ void async_copy32B(unsigned lds_off,
                                                     const _Float16* g) {
  unsigned long long ga = (unsigned long long)(uintptr_t)g;
  asm volatile("global_load_async_to_lds_b128 %0, %1, off\n\t"
               "global_load_async_to_lds_b128 %0, %1, off offset:16"
               :: "v"(lds_off), "v"(ga) : "memory");
}
static __device__ __forceinline__ void wait_async_le8() {
  asm volatile("s_wait_asynccnt 0x8" ::: "memory");
}
static __device__ __forceinline__ void wait_async_0() {
  asm volatile("s_wait_asynccnt 0x0" ::: "memory");
}

// ---------------------------------------------------------------------------
// Kernel 0a: split f32 -> f16 hi + f16 lo residual (error-compensated pair)
// ---------------------------------------------------------------------------
__global__ void split_kernel(const float* __restrict__ x,
                             _Float16* __restrict__ hi,
                             _Float16* __restrict__ lo, int n) {
  for (int i = blockIdx.x * blockDim.x + threadIdx.x; i < n;
       i += gridDim.x * blockDim.x) {
    float v = x[i];
    _Float16 h = (_Float16)v;
    hi[i] = h;
    lo[i] = (_Float16)(v - (float)h);
  }
}

// Kernel 0b: W_a [c][q] -> WaT [q][c], split hi/lo (contiguous-K B fragments)
__global__ void wa_transpose_split(const float* __restrict__ Wa,
                                   _Float16* __restrict__ hi,
                                   _Float16* __restrict__ lo) {
  int i = blockIdx.x * blockDim.x + threadIdx.x;
  if (i >= DC * DQ) return;
  int qi = i >> 10, c = i & 1023;
  float v = Wa[(size_t)c * DQ + qi];
  _Float16 h = (_Float16)v;
  hi[i] = h;
  lo[i] = (_Float16)(v - (float)h);
}

// Kernel 0c: W_p [100][1024] -> padded [112][1024], split hi/lo
__global__ void wp_pad_split(const float* __restrict__ Wp,
                             _Float16* __restrict__ hi,
                             _Float16* __restrict__ lo) {
  int i = blockIdx.x * blockDim.x + threadIdx.x;
  if (i >= PPAD * DC) return;
  int p = i >> 10, c = i & 1023;
  float v = (p < PSIZE) ? Wp[(size_t)p * DC + c] : 0.f;
  _Float16 h = (_Float16)v;
  hi[i] = h;
  lo[i] = (_Float16)(v - (float)h);
}

// ---------------------------------------------------------------------------
// Kernel 1: p_t = S * sigmoid( V_p . tanh(c_t @ W_p^T) )
// One wave per 16-row tile; 7 column tiles (P padded to 112); 3-term f16 WMMA.
// ---------------------------------------------------------------------------
__global__ __launch_bounds__(256) void pt_kernel(
    const _Float16* __restrict__ cHi, const _Float16* __restrict__ cLo,
    const _Float16* __restrict__ wpHi, const _Float16* __restrict__ wpLo,
    const float* __restrict__ Vp, float* __restrict__ pt) {
  const int tid  = threadIdx.x;
  const int wave = tid >> 5;
  const int lane = tid & 31;
  const int l15  = lane & 15;
  const int kb   = (lane >> 4) * 8;
  const int rowBase = (blockIdx.x * 8 + wave) * 16;

  v8f acc[7];
  for (int ct = 0; ct < 7; ++ct)
    for (int r = 0; r < 8; ++r) acc[ct][r] = 0.f;

  const _Float16* aHiP = cHi + (size_t)(rowBase + l15) * DC;
  const _Float16* aLoP = cLo + (size_t)(rowBase + l15) * DC;

  for (int k0 = 0; k0 < DC; k0 += 32) {
    Frag ah, al;
    ah.u[0] = *(const uint4*)(aHiP + k0 + kb);        // halves K[kb..kb+7]
    ah.u[1] = *(const uint4*)(aHiP + k0 + kb + 16);   // halves K[16+kb..+7]
    al.u[0] = *(const uint4*)(aLoP + k0 + kb);
    al.u[1] = *(const uint4*)(aLoP + k0 + kb + 16);
    for (int ct = 0; ct < 7; ++ct) {
      const _Float16* bHiP = wpHi + (size_t)(ct * 16 + l15) * DC + k0;
      const _Float16* bLoP = wpLo + (size_t)(ct * 16 + l15) * DC + k0;
      Frag bh, bl;
      bh.u[0] = *(const uint4*)(bHiP + kb);
      bh.u[1] = *(const uint4*)(bHiP + kb + 16);
      bl.u[0] = *(const uint4*)(bLoP + kb);
      bl.u[1] = *(const uint4*)(bLoP + kb + 16);
      acc[ct] = wmma_f16(ah.h, bh.h, acc[ct]);
      acc[ct] = wmma_f16(ah.h, bl.h, acc[ct]);
      acc[ct] = wmma_f16(al.h, bh.h, acc[ct]);
    }
  }

  // tanh + dot with V_p; C layout: row = r + 8*(lane>>4), p = ct*16 + (lane&15)
  float part[8];
  for (int r = 0; r < 8; ++r) part[r] = 0.f;
  for (int ct = 0; ct < 7; ++ct) {
    int p = ct * 16 + l15;
    float vp = (p < PSIZE) ? Vp[p] : 0.f;
    for (int r = 0; r < 8; ++r) part[r] += vp * tanhf(acc[ct][r]);
  }
  // reduce across the 16 lanes of each half-wave (wave32: width 32, masks 1..8)
  for (int m = 1; m <= 8; m <<= 1)
    for (int r = 0; r < 8; ++r) part[r] += __shfl_xor(part[r], m, 32);
  if (l15 == 0) {
    int rb = rowBase + (lane >> 4) * 8;
    for (int r = 0; r < 8; ++r)
      pt[rb + r] = (float)S_LEN / (1.f + __expf(-part[r]));
  }
}

// ---------------------------------------------------------------------------
// Kernel 2: u[8192x1024] = c_t @ W_a  via hi/lo split f16 WMMA.
// 128x128 WG tile, BK=32; async global->LDS double buffering (ASYNCcnt).
// ---------------------------------------------------------------------------
#define ARR   (128 * LDSTRIDE)   /* halves per (array, buffer) */
#define BUFSZ (4 * ARR)          /* Ah, Al, Bh, Bl */

__global__ __launch_bounds__(256) void gemm_kernel(
    const _Float16* __restrict__ aHi, const _Float16* __restrict__ aLo,
    const _Float16* __restrict__ bHi, const _Float16* __restrict__ bLo,
    float* __restrict__ U) {
  __shared__ _Float16 smem[2 * BUFSZ];

  const int tid  = threadIdx.x;
  const int wave = tid >> 5;
  const int lane = tid & 31;
  const int l15  = lane & 15;
  const int kb   = (lane >> 4) * 8;
  const int rowBase = blockIdx.x * 128;
  const int colBase = blockIdx.y * 128;

  // staging map: thread -> (row 0..127, 16-half chunk 0/1); 32B/thread/array
  const int lr = tid >> 1;
  const int lc = (tid & 1) * 16;
  const _Float16* gA0 = aHi + (size_t)(rowBase + lr) * DC + lc;
  const _Float16* gA1 = aLo + (size_t)(rowBase + lr) * DC + lc;
  const _Float16* gB0 = bHi + (size_t)(colBase + lr) * DC + lc;
  const _Float16* gB1 = bLo + (size_t)(colBase + lr) * DC + lc;

  // LDS byte offset of this thread's staging slot (generic ptr low 32 bits)
  const unsigned ldsBase =
      (unsigned)(uintptr_t)&smem[lr * LDSTRIDE + lc];

  v8f acc[8];
  for (int ct = 0; ct < 8; ++ct)
    for (int r = 0; r < 8; ++r) acc[ct][r] = 0.f;

  // prologue: async-copy K-panel 0 into buffer 0 (8 b128 per thread)
  async_copy32B(ldsBase + 0 * ARR * 2, gA0);
  async_copy32B(ldsBase + 1 * ARR * 2, gA1);
  async_copy32B(ldsBase + 2 * ARR * 2, gB0);
  async_copy32B(ldsBase + 3 * ARR * 2, gB1);

  for (int s = 0; s < DC / 32; ++s) {
    const int cur = s & 1;
    if (s + 1 < DC / 32) {
      // kick off next panel into the other buffer, then wait for current
      const int k1 = (s + 1) * 32;
      const unsigned nb = ldsBase + (unsigned)((1 - cur) * BUFSZ) * 2;
      async_copy32B(nb + 0 * ARR * 2, gA0 + k1);
      async_copy32B(nb + 1 * ARR * 2, gA1 + k1);
      async_copy32B(nb + 2 * ARR * 2, gB0 + k1);
      async_copy32B(nb + 3 * ARR * 2, gB1 + k1);
      wait_async_le8();   // newest 8 = next panel; older (current) complete
    } else {
      wait_async_0();
    }
    __syncthreads();

    const _Float16* sAh = &smem[cur * BUFSZ + 0 * ARR];
    const _Float16* sAl = &smem[cur * BUFSZ + 1 * ARR];
    const _Float16* sBh = &smem[cur * BUFSZ + 2 * ARR];
    const _Float16* sBl = &smem[cur * BUFSZ + 3 * ARR];

    Frag ah, al;
    const int ar = (wave * 16 + l15) * LDSTRIDE;
    ah.u[0] = *(const uint4*)&sAh[ar + kb];
    ah.u[1] = *(const uint4*)&sAh[ar + kb + 16];
    al.u[0] = *(const uint4*)&sAl[ar + kb];
    al.u[1] = *(const uint4*)&sAl[ar + kb + 16];
    for (int ct = 0; ct < 8; ++ct) {
      const int br = (ct * 16 + l15) * LDSTRIDE;
      Frag bh, bl;
      bh.u[0] = *(const uint4*)&sBh[br + kb];
      bh.u[1] = *(const uint4*)&sBh[br + kb + 16];
      bl.u[0] = *(const uint4*)&sBl[br + kb];
      bl.u[1] = *(const uint4*)&sBl[br + kb + 16];
      acc[ct] = wmma_f16(ah.h, bh.h, acc[ct]);
      acc[ct] = wmma_f16(ah.h, bl.h, acc[ct]);
      acc[ct] = wmma_f16(al.h, bh.h, acc[ct]);
    }
    __syncthreads();  // everyone done reading buf[cur] before it is refilled
  }

  const int orow = rowBase + wave * 16 + (lane >> 4) * 8;
  const int ocol = colBase + l15;
  for (int ct = 0; ct < 8; ++ct)
    for (int r = 0; r < 8; ++r)
      U[(size_t)(orow + r) * DQ + ocol + ct * 16] = acc[ct][r];
}

// ---------------------------------------------------------------------------
// Kernel 3: per-row window gather + scores + softmax*gauss + weighted sum
// ---------------------------------------------------------------------------
__global__ __launch_bounds__(256) void attn_kernel(
    const float* __restrict__ q, const float* __restrict__ U,
    const float* __restrict__ pt, float* __restrict__ out) {
  const int row  = blockIdx.x;        // 0..8191 = b*T + t
  const int b    = row >> 11;         // T = 2048
  const int tid  = threadIdx.x;
  const int wave = tid >> 5;
  const int lane = tid & 31;

  const float pr = pt[row];
  const int wl = (int)pr - 2;         // trunc == floor (pr >= 0)
  int wi[4];
  for (int w = 0; w < 4; ++w) {
    int x = wl + w;
    wi[w] = x < 0 ? 0 : (x > S_LEN - 1 ? S_LEN - 1 : x);
  }

  const float* uRow = U + (size_t)row * DQ;
  const float* qB   = q + (size_t)b * S_LEN * DQ;

  float qv[4][4];
  float s[4] = {0.f, 0.f, 0.f, 0.f};
  for (int j = 0; j < 4; ++j) {
    int col = tid + j * 256;
    float uv = uRow[col];
    for (int w = 0; w < 4; ++w) {
      float qq = qB[(size_t)wi[w] * DQ + col];
      qv[w][j] = qq;
      s[w] += uv * qq;
    }
  }
  for (int off = 16; off; off >>= 1)
    for (int w = 0; w < 4; ++w) s[w] += __shfl_down(s[w], off, 32);

  __shared__ float red[8][4];
  __shared__ float sa[4];
  if (lane == 0)
    for (int w = 0; w < 4; ++w) red[wave][w] = s[w];
  __syncthreads();
  if (tid == 0) {
    float sc[4];
    for (int w = 0; w < 4; ++w) {
      float t = 0.f;
      for (int k = 0; k < 8; ++k) t += red[k][w];
      sc[w] = t;
    }
    float m = sc[0];
    for (int w = 1; w < 4; ++w) m = fmaxf(m, sc[w]);
    float e[4], se = 0.f;
    for (int w = 0; w < 4; ++w) { e[w] = __expf(sc[w] - m); se += e[w]; }
    for (int w = 0; w < 4; ++w) {
      float d = (float)wi[w] - pr;
      sa[w] = (e[w] / se) * __expf(-0.5f * d * d);  // STAN_DEV = 1.0
    }
  }
  __syncthreads();
  const float a0 = sa[0], a1 = sa[1], a2 = sa[2], a3 = sa[3];
  float* oRow = out + (size_t)row * DQ;
  for (int j = 0; j < 4; ++j) {
    int col = tid + j * 256;
    oRow[col] = a0 * qv[0][j] + a1 * qv[1][j] + a2 * qv[2][j] + a3 * qv[3][j];
  }
}

// ---------------------------------------------------------------------------
extern "C" void kernel_launch(void* const* d_in, const int* in_sizes, int n_in,
                              void* d_out, int out_size, void* d_ws, size_t ws_size,
                              hipStream_t stream) {
  (void)in_sizes; (void)n_in; (void)out_size; (void)ws_size;
  const float* q  = (const float*)d_in[0];
  const float* ct = (const float*)d_in[1];
  const float* Wa = (const float*)d_in[2];
  const float* Wp = (const float*)d_in[3];
  const float* Vp = (const float*)d_in[4];
  float* out = (float*)d_out;

  char* ws = (char*)d_ws;
  size_t off = 0;
  auto wsAlloc = [&](size_t bytes) -> void* {
    void* p = ws + off;
    off += (bytes + 255) & ~(size_t)255;
    return p;
  };
  _Float16* cHi  = (_Float16*)wsAlloc((size_t)NROWS * DC * 2);
  _Float16* cLo  = (_Float16*)wsAlloc((size_t)NROWS * DC * 2);
  _Float16* waHi = (_Float16*)wsAlloc((size_t)DQ * DC * 2);
  _Float16* waLo = (_Float16*)wsAlloc((size_t)DQ * DC * 2);
  _Float16* wpHi = (_Float16*)wsAlloc((size_t)PPAD * DC * 2);
  _Float16* wpLo = (_Float16*)wsAlloc((size_t)PPAD * DC * 2);
  float*    ptb  = (float*)wsAlloc((size_t)NROWS * 4);
  float*    Ubuf = (float*)wsAlloc((size_t)NROWS * DQ * 4);

  split_kernel<<<4096, 256, 0, stream>>>(ct, cHi, cLo, NROWS * DC);
  wa_transpose_split<<<(DQ * DC) / 256, 256, 0, stream>>>(Wa, waHi, waLo);
  wp_pad_split<<<(PPAD * DC) / 256, 256, 0, stream>>>(Wp, wpHi, wpLo);
  pt_kernel<<<NROWS / 128, 256, 0, stream>>>(cHi, cLo, wpHi, wpLo, Vp, ptb);
  gemm_kernel<<<dim3(NROWS / 128, DQ / 128), 256, 0, stream>>>(cHi, cLo, waHi, waLo, Ubuf);
  attn_kernel<<<NROWS, 256, 0, stream>>>(q, Ubuf, ptb, out);
}
